// PetsGraphSAGE_11905649344801
// MI455X (gfx1250) — matrix operand order repaired and checked by reading, not compiled, not run on hardware
//
#include <hip/hip_runtime.h>

typedef __attribute__((ext_vector_type(16))) _Float16 v16h;
typedef __attribute__((ext_vector_type(8)))  float    v8f;
typedef __attribute__((ext_vector_type(4)))  unsigned int v4u;
typedef __attribute__((ext_vector_type(8)))  unsigned int v8u;

#define HID 128
#define ASTRIDE 256   // [h1 | mean] concatenated A matrix, f16, row stride 256

// ---------------- 1) zero init: deg, agg1, agg2 region of wsA -------------
__global__ __launch_bounds__(256) void zero_init(float* deg, float* agg1,
                                                 _Float16* wsA, int N) {
  long i = (long)blockIdx.x * 256 + threadIdx.x;
  long total = (long)N * 64;                 // agg2 region as u32 words
  if (i < total) {
    int n = (int)(i >> 6), c = (int)(i & 63);
    ((unsigned int*)(wsA + (long)n * ASTRIDE + HID))[c] = 0u;
  }
  if (i < N) deg[i] = 0.f;
  if (i < 4L * N) agg1[i] = 0.f;
}

// ---------------- 2) edge pass 1: degree + feat scatter (fp32, d=4) -------
__global__ __launch_bounds__(256) void edge_l1(const int* __restrict__ src,
                                               const int* __restrict__ dst,
                                               const float* __restrict__ feats,
                                               float* deg, float* agg1, int E) {
  int e = blockIdx.x * 256 + threadIdx.x;
  if (e >= E) return;
  int s = src[e], d = dst[e];
  float4 f = ((const float4*)feats)[s];
  atomicAdd(&deg[d], 1.f);
  float* a = agg1 + (long)d * 4;
  atomicAdd(a + 0, f.x); atomicAdd(a + 1, f.y);
  atomicAdd(a + 2, f.z); atomicAdd(a + 3, f.w);
}

// ---------------- 3) L1 dense (K=4 -> scalar FMA), write h1 as f16 --------
__global__ __launch_bounds__(256) void l1_compute(const float* __restrict__ feats,
                                                  const float* __restrict__ agg1,
                                                  const float* __restrict__ deg,
                                                  const float* __restrict__ ws1,
                                                  const float* __restrict__ wn1,
                                                  const float* __restrict__ b1,
                                                  _Float16* __restrict__ wsA, int N) {
  long i = (long)blockIdx.x * 256 + threadIdx.x;
  if (i >= (long)N * HID) return;
  int n = (int)(i >> 7), j = (int)(i & 127);
  float d = deg[n]; d = d > 1.f ? d : 1.f;
  float inv = 1.f / d;
  const float* f = feats + (long)n * 4;
  const float* a = agg1 + (long)n * 4;
  float acc = b1[j];
#pragma unroll
  for (int k = 0; k < 4; ++k)
    acc += f[k] * ws1[k * HID + j] + (a[k] * inv) * wn1[k * HID + j];
  acc = fmaxf(acc, 0.f);
  wsA[(long)n * ASTRIDE + j] = (_Float16)acc;
}

// ---------------- 4) edge pass 2: wave per edge, pk f16 atomic scatter ----
__global__ __launch_bounds__(256) void edge_l2(const int* __restrict__ src,
                                               const int* __restrict__ dst,
                                               _Float16* __restrict__ wsA, int E) {
  int gid = blockIdx.x * 256 + threadIdx.x;
  int e = gid >> 5;
  int lane = threadIdx.x & 31;
  if (e >= E) return;
  int s = src[e], d = dst[e];
  // gather 8B (4 halves) per lane, coalesced 256B per edge row
  uint2 v = ((const uint2*)(wsA + (long)s * ASTRIDE))[lane];
  _Float16* drow = wsA + (long)d * ASTRIDE + HID + lane * 4;
  unsigned long long a0 = (unsigned long long)drow;
  unsigned long long a1 = a0 + 4;
  // CDNA5 packed f16 atomic add, no return (STOREcnt-tracked)
  asm volatile("global_atomic_pk_add_f16 %0, %1, off" :: "v"(a0), "v"(v.x) : "memory");
  asm volatile("global_atomic_pk_add_f16 %0, %1, off" :: "v"(a1), "v"(v.y) : "memory");
}

// ---------------- 5) normalize agg2 by degree -----------------------------
__global__ __launch_bounds__(256) void l2_normalize(_Float16* __restrict__ wsA,
                                                    const float* __restrict__ deg, int N) {
  long i = (long)blockIdx.x * 256 + threadIdx.x;
  if (i >= (long)N * 64) return;
  int n = (int)(i >> 6), c = (int)(i & 63);
  _Float16* p = wsA + (long)n * ASTRIDE + HID + c * 2;
  float d = deg[n]; d = d > 1.f ? d : 1.f;
  float inv = 1.f / d;
  p[0] = (_Float16)((float)p[0] * inv);
  p[1] = (_Float16)((float)p[1] * inv);
}

// ---------------- 6) pack [w_self2; w_neigh2] -> f16 Wcat [256][128] ------
__global__ __launch_bounds__(256) void build_wcat(const float* __restrict__ ws2,
                                                  const float* __restrict__ wn2,
                                                  _Float16* __restrict__ Wcat) {
  int i = blockIdx.x * 256 + threadIdx.x;
  if (i >= 2 * HID * HID) return;
  int k = i >> 7, n = i & 127;
  float w = (k < HID) ? ws2[k * HID + n] : wn2[(k - HID) * HID + n];
  Wcat[i] = (_Float16)w;
}

// ---------------- 7) WMMA GEMM: h2 = relu([h1|mean] @ Wcat + b2) ----------
// block = 256 threads = 8 waves; block tile 128 rows x 128 cols.
// Wave w owns rows [row0+16w, +16), all 128 cols as 8 C fragments.
// Wcat staged into LDS by the Tensor Data Mover, one 128-row K-half at a
// time.  TDM pad fields give the bank-conflict padding for free:
//   rows are 64 dwords (pad_interval code 5 = 64 dwords), pad_amount code 3
//   (= 4 dwords) -> LDS row stride 136 halves (68 dwords, <=2-way conflict,
//   16B aligned).
__global__ __launch_bounds__(256) void l2_gemm(const _Float16* __restrict__ A,
                                               const _Float16* __restrict__ Wcat,
                                               const float* __restrict__ b2,
                                               float* __restrict__ h2, int N) {
  __shared__ _Float16 Wlds[128][136];
  int tid = threadIdx.x;
  int wave = tid >> 5, lane = tid & 31;
  int row0 = blockIdx.x * 128 + wave * 16;

  // A fragment addressing (16-bit A 16x32 layout):
  // lane m=lane&15, k-offset = (lane>>4)*8 ; two 16B chunks at +0 and +16 halves
  int m_ld = lane & 15;
  int koff = (lane >> 4) << 3;
  int arow = row0 + m_ld; if (arow >= N) arow = N - 1;   // clamp OOB reads
  const _Float16* abase = A + (long)arow * ASTRIDE;

  v8f c[8];
  v8f zz = {};
#pragma unroll
  for (int t = 0; t < 8; ++t) c[t] = zz;

  for (int half = 0; half < 2; ++half) {
    __syncthreads();          // previous half's B reads complete
    if (wave == 0) {
      // ---- TDM descriptor (D#), 2D tensor: 64 dwords x 128 rows ----
      unsigned ldsbase = (unsigned)(unsigned long long)(&Wlds[0][0]);
      unsigned long long ga =
          (unsigned long long)(Wcat + (long)half * HID * HID);
      v4u g0;
      g0.x = 1u;                                   // count=1, user mode
      g0.y = ldsbase;                              // lds_addr
      g0.z = (unsigned)ga;                         // global_addr[31:0]
      g0.w = (unsigned)(ga >> 32) | (2u << 30);    // addr[56:32] | type=2
      v8u g1;
      g1.s0 = (2u << 16)      // data_size = 4 bytes
            | (1u << 20)      // pad_enable
            | (5u << 22)      // pad_interval: 64 dwords
            | (3u << 25);     // pad_amount:   4 dwords
      g1.s1 = (64u << 16);    // tensor_dim0 = 64 dwords (lo16 in bits 63:48)
      g1.s2 = (128u << 16);   // tensor_dim0 hi=0 | tensor_dim1 = 128 rows
      g1.s3 = (64u << 16);    // tensor_dim1 hi=0 | tile_dim0 = 64 dwords
      g1.s4 = 128u;           // tile_dim1 = 128 rows | tile_dim2 = 0
      g1.s5 = 64u;            // tensor_dim0_stride = 64 dwords
      g1.s6 = 0u;             // stride hi | tensor_dim1_stride lo
      g1.s7 = 0u;             // tensor_dim1_stride hi
      asm volatile("tensor_load_to_lds %0, %1" :: "s"(g0), "s"(g1) : "memory");
      __builtin_amdgcn_s_wait_tensorcnt(0);
    }
    __syncthreads();          // LDS tile visible to all 8 waves

#pragma unroll
    for (int k0 = 0; k0 < 128; k0 += 32) {
      union { v16h v; uint4 q[2]; } a;
      const _Float16* ap = abase + half * 128 + k0 + koff;
      a.q[0] = *(const uint4*)(ap);
      a.q[1] = *(const uint4*)(ap + 16);
      int krow = k0 + koff;        // B: lane L holds col n=L&15, k=k0+koff+s (s<8), k0+16+koff+s-8 (s>=8)
#pragma unroll
      for (int t = 0; t < 8; ++t) {
        union { v16h v; uint4 q[2]; } b;
        const _Float16* bp = &Wlds[krow][t * 16 + m_ld];
#pragma unroll
        for (int s = 0; s < 8; ++s) {
          b.v[s] = bp[s * 136];
          b.v[s + 8] = bp[(16 + s) * 136];
        }
        c[t] = __builtin_amdgcn_wmma_f32_16x16x32_f16(
            false, a.v, false, b.v, (short)0, c[t], false, false);
      }
    }
  }

  // C layout: VGPR i, lanes 0-15 -> M=i ; lanes 16-31 -> M=i+8 ; N = lane&15
  int mhi = (lane >> 4) << 3;
  int n = lane & 15;
#pragma unroll
  for (int i = 0; i < 8; ++i) {
    int row = row0 + i + mhi;
    if (row < N) {
#pragma unroll
      for (int t = 0; t < 8; ++t) {
        float v = c[t][i] + b2[t * 16 + n];
        h2[(long)row * HID + t * 16 + n] = fmaxf(v, 0.f);
      }
    }
  }
}

// ---------------- 8) output logits: wave per node, shfl reduce ------------
__global__ __launch_bounds__(256) void out_logits(const float* __restrict__ h2,
                                                  const float* __restrict__ wout,
                                                  const float* __restrict__ bout,
                                                  float* __restrict__ out, int N) {
  int gid = blockIdx.x * 256 + threadIdx.x;
  int node = gid >> 5;
  int lane = threadIdx.x & 31;
  if (node >= N) return;
  const float* hr = h2 + (long)node * HID;
  float a0 = 0.f, a1 = 0.f;
#pragma unroll
  for (int t = 0; t < 4; ++t) {
    int k = lane + t * 32;
    float v = hr[k];
    a0 += v * wout[k * 2 + 0];
    a1 += v * wout[k * 2 + 1];
  }
#pragma unroll
  for (int off = 16; off > 0; off >>= 1) {
    a0 += __shfl_down(a0, off, 32);
    a1 += __shfl_down(a1, off, 32);
  }
  if (lane == 0) {
    out[(long)node * 2 + 0] = a0 + bout[0];
    out[(long)node * 2 + 1] = a1 + bout[1];
  }
}

extern "C" void kernel_launch(void* const* d_in, const int* in_sizes, int n_in,
                              void* d_out, int out_size, void* d_ws, size_t ws_size,
                              hipStream_t stream) {
  const float* feats = (const float*)d_in[0];
  const int*   src   = (const int*)d_in[1];
  const int*   dst   = (const int*)d_in[2];
  const float* ws1   = (const float*)d_in[3];
  const float* wn1   = (const float*)d_in[4];
  const float* b1    = (const float*)d_in[5];
  const float* ws2   = (const float*)d_in[6];
  const float* wn2   = (const float*)d_in[7];
  const float* b2    = (const float*)d_in[8];
  const float* wout  = (const float*)d_in[9];
  const float* bout  = (const float*)d_in[10];
  float* out = (float*)d_out;

  const int N = in_sizes[0] / 4;   // feats [N,4]
  const int E = in_sizes[1];

  // workspace layout (256B aligned slices)
  size_t off = 0;
  auto take = [&](size_t bytes) {
    size_t o = off;
    off = (off + bytes + 255) & ~(size_t)255;
    return o;
  };
  char* w = (char*)d_ws;
  float*    deg  = (float*)(w + take((size_t)N * 4));
  float*    agg1 = (float*)(w + take((size_t)N * 16));
  _Float16* wsA  = (_Float16*)(w + take((size_t)N * ASTRIDE * 2 + 65536));
  _Float16* Wcat = (_Float16*)(w + take((size_t)2 * HID * HID * 2));
  float*    h2   = (float*)(w + take((size_t)N * HID * 4));

  long zelems = (long)N * 64;
  zero_init<<<(int)((zelems + 255) / 256), 256, 0, stream>>>(deg, agg1, wsA, N);
  edge_l1<<<(E + 255) / 256, 256, 0, stream>>>(src, dst, feats, deg, agg1, E);
  long l1e = (long)N * HID;
  l1_compute<<<(int)((l1e + 255) / 256), 256, 0, stream>>>(feats, agg1, deg, ws1, wn1, b1, wsA, N);
  build_wcat<<<(2 * HID * HID + 255) / 256, 256, 0, stream>>>(ws2, wn2, Wcat);
  edge_l2<<<(int)(((long)E * 32 + 255) / 256), 256, 0, stream>>>(src, dst, wsA, E);
  l2_normalize<<<(int)((zelems + 255) / 256), 256, 0, stream>>>(wsA, deg, N);
  l2_gemm<<<(N + 127) / 128, 256, 0, stream>>>(wsA, Wcat, b2, h2, N);
  out_logits<<<(int)(((long)N * 32 + 255) / 256), 256, 0, stream>>>(h2, wout, bout, out, N);
}
